// GraphConvolutionSparse_44014824849453
// MI455X (gfx1250) — compile-verified, matrix operand order checked by reading, not run on hardware
//
#include <hip/hip_runtime.h>
#include <hip/hip_bf16.h>
#include <stdint.h>

#define NODES   100000
#define IN_DIM  512
#define OUT_DIM 256

typedef __attribute__((ext_vector_type(16))) __bf16 v16bf;
typedef __attribute__((ext_vector_type(8)))  float  v8f;

// ---------------------------------------------------------------------------
// f32 -> bf16 round-to-nearest-even (bit trick)
// ---------------------------------------------------------------------------
__device__ __forceinline__ unsigned short f32_to_bf16(float f) {
    union { float f; uint32_t u; } cv; cv.f = f;
    uint32_t u = cv.u + 0x7FFFu + ((cv.u >> 16) & 1u);
    return (unsigned short)(u >> 16);
}

// Build a 16-element bf16 fragment from two 16-byte runs (K and K+16 halves).
__device__ __forceinline__ v16bf frag_from_2x16B(uint4 lo, uint4 hi) {
    union { uint4 q[2]; v16bf v; } u;
    u.q[0] = lo; u.q[1] = hi;
    return u.v;
}

// ---------------------------------------------------------------------------
// Phase 0: zero a region (uint4 granularity)
// ---------------------------------------------------------------------------
__global__ void zero_ws_kernel(uint4* __restrict__ p, long long n4) {
    long long i = (long long)blockIdx.x * blockDim.x + threadIdx.x;
    if (i < n4) p[i] = uint4{0u, 0u, 0u, 0u};
}

// ---------------------------------------------------------------------------
// Phase 1: Wtb[n][k] = bf16(W[k][n])  (transpose + convert once)
// ---------------------------------------------------------------------------
__global__ void transpose_w_kernel(const float* __restrict__ W,
                                   unsigned short* __restrict__ Wtb) {
    int i = blockIdx.x * blockDim.x + threadIdx.x;   // over IN_DIM*OUT_DIM
    if (i < IN_DIM * OUT_DIM) {
        int k = i >> 8;          // row of W
        int n = i & (OUT_DIM-1); // col of W
        Wtb[(long long)n * IN_DIM + k] = f32_to_bf16(W[i]);
    }
}

// ---------------------------------------------------------------------------
// Phase 2: densify sparse features: X[r][c] += v  (3.2M f32 atomics, cheap)
// ---------------------------------------------------------------------------
__global__ void scatter_feat_kernel(const int* __restrict__ rows,
                                    const int* __restrict__ cols,
                                    const float* __restrict__ vals,
                                    float* __restrict__ X, int nnz) {
    int i = blockIdx.x * blockDim.x + threadIdx.x;
    if (i >= nnz) return;
    atomicAdd(X + (long long)rows[i] * IN_DIM + cols[i], vals[i]);
}

// ---------------------------------------------------------------------------
// Phase 3: Xb = bf16(X), 8 elements per thread (2x b128 in, 1x b128 out)
// ---------------------------------------------------------------------------
__global__ void convert_x_bf16_kernel(const float* __restrict__ X,
                                      uint4* __restrict__ Xb, long long n8) {
    long long i = (long long)blockIdx.x * blockDim.x + threadIdx.x;
    if (i >= n8) return;
    const float4* p = (const float4*)(X + i * 8);
    float4 a = p[0], b = p[1];
    union { unsigned short s[8]; uint4 q; } u;
    u.s[0] = f32_to_bf16(a.x); u.s[1] = f32_to_bf16(a.y);
    u.s[2] = f32_to_bf16(a.z); u.s[3] = f32_to_bf16(a.w);
    u.s[4] = f32_to_bf16(b.x); u.s[5] = f32_to_bf16(b.y);
    u.s[6] = f32_to_bf16(b.z); u.s[7] = f32_to_bf16(b.w);
    Xb[i] = u.q;
}

// ---------------------------------------------------------------------------
// Phase 4: H = Xb @ W  (bf16 WMMA, f32 accumulate)
// 8-wave workgroup computes a 32-row x 256-col block (grid = 100000/32 = 3125).
// The A block (32 x 512 bf16 = 32 KB, contiguous) is staged into LDS with
// global_load_async_to_lds_b128; fragments then come from LDS (ds_load_b128).
// Each wave owns two m-tiles x two n-tiles: every B fragment (global b128 from
// the L2-resident Wtb) feeds two WMMAs -> 4 WMMAs per k-step per wave.
// ---------------------------------------------------------------------------
__global__ __launch_bounds__(256)
void spmm1_wmma_kernel(const unsigned short* __restrict__ Xb,
                       const unsigned short* __restrict__ Wtb,
                       float* __restrict__ H) {
    __shared__ unsigned short Atile[32 * IN_DIM];   // 32 KB

    const int tid   = threadIdx.x;
    const int wave  = tid >> 5;
    const int lane  = tid & 31;
    const int m0    = blockIdx.x * 32;
    const int mrow  = lane & 15;
    const int khalf = (lane >> 4) * 8;          // 0 or 8 per A/B layout
    const int n0    = wave * 32;

    // --- async copy of the contiguous 32x512 bf16 A block into LDS ---------
    {
        const unsigned lds_base = (unsigned)(size_t)&Atile[0];
        const uint64_t gbase = (uint64_t)(Xb + (size_t)m0 * IN_DIM);
        #pragma unroll
        for (int it = 0; it < 8; ++it) {
            unsigned loff = lds_base + (unsigned)(tid * 16 + it * 4096);
            uint64_t ga   = gbase + (uint64_t)(tid * 16 + it * 4096);
            asm volatile("global_load_async_to_lds_b128 %0, %1, off"
                         :: "v"(loff), "v"(ga) : "memory");
        }
        asm volatile("s_wait_asynccnt 0" ::: "memory");
        __syncthreads();
    }

    const unsigned short* a0row = &Atile[mrow * IN_DIM + khalf];
    const unsigned short* a1row = a0row + 16 * IN_DIM;
    const unsigned short* b0row = Wtb + (size_t)(n0 + mrow)      * IN_DIM + khalf;
    const unsigned short* b1row = Wtb + (size_t)(n0 + 16 + mrow) * IN_DIM + khalf;

    v8f c00 = {}, c01 = {}, c10 = {}, c11 = {};
    #pragma unroll 2
    for (int kk = 0; kk < IN_DIM; kk += 32) {
        v16bf a0 = frag_from_2x16B(*(const uint4*)(a0row + kk),
                                   *(const uint4*)(a0row + kk + 16));
        v16bf a1 = frag_from_2x16B(*(const uint4*)(a1row + kk),
                                   *(const uint4*)(a1row + kk + 16));
        v16bf b0 = frag_from_2x16B(*(const uint4*)(b0row + kk),
                                   *(const uint4*)(b0row + kk + 16));
        v16bf b1 = frag_from_2x16B(*(const uint4*)(b1row + kk),
                                   *(const uint4*)(b1row + kk + 16));
        c00 = __builtin_amdgcn_wmma_f32_16x16x32_bf16(false, a0, false, b0,
                                                      (short)0, c00, false, false);
        c10 = __builtin_amdgcn_wmma_f32_16x16x32_bf16(false, a1, false, b0,
                                                      (short)0, c10, false, false);
        c01 = __builtin_amdgcn_wmma_f32_16x16x32_bf16(false, a0, false, b1,
                                                      (short)0, c01, false, false);
        c11 = __builtin_amdgcn_wmma_f32_16x16x32_bf16(false, a1, false, b1,
                                                      (short)0, c11, false, false);
    }

    // C/D layout: VGPR j -> row mtile0 + j + 8*(lane>>4), col n + (lane&15)
    const int rbase = m0 + (lane >> 4) * 8;
    float* hp0 = H + (size_t)rbase * OUT_DIM;          // m-tile 0
    float* hp1 = hp0 + (size_t)16 * OUT_DIM;           // m-tile 1
    #pragma unroll
    for (int j = 0; j < 8; ++j) {
        const size_t ro = (size_t)j * OUT_DIM;
        hp0[ro + n0      + (lane & 15)] = c00[j];
        hp0[ro + n0 + 16 + (lane & 15)] = c01[j];
        hp1[ro + n0      + (lane & 15)] = c10[j];
        hp1[ro + n0 + 16 + (lane & 15)] = c11[j];
    }
}

// ---------------------------------------------------------------------------
// Phase 5: acc[r] += val * H[c]  (one wave per edge; gathers are L2-resident)
// ---------------------------------------------------------------------------
__global__ __launch_bounds__(256)
void scatter_edges_kernel(const int* __restrict__ rows,
                          const int* __restrict__ cols,
                          const float* __restrict__ vals,
                          const float* __restrict__ H,
                          float* __restrict__ acc, int nedges) {
    int e = blockIdx.x * 8 + (threadIdx.x >> 5);
    if (e >= nedges) return;
    const int lane = threadIdx.x & 31;
    const int   r = rows[e];
    const int   c = cols[e];
    const float v = vals[e];
    const float* hs = H   + (size_t)c * OUT_DIM;
    float*       dp = acc + (size_t)r * OUT_DIM;
    __builtin_prefetch(hs + lane, 0, 0);
    #pragma unroll
    for (int j = 0; j < OUT_DIM; j += 32)
        atomicAdd(dp + j + lane, v * hs[j + lane]);
}

// ---------------------------------------------------------------------------
// Phase 6: out = relu(acc)
// ---------------------------------------------------------------------------
__global__ void relu_kernel(const float* __restrict__ acc, float* __restrict__ out,
                            long long n) {
    long long i = (long long)blockIdx.x * blockDim.x + threadIdx.x;
    if (i < n) out[i] = fmaxf(acc[i], 0.0f);
}

// ---------------------------------------------------------------------------
// Host-side launch sequence (graph-capture safe: plain launches only)
// ---------------------------------------------------------------------------
extern "C" void kernel_launch(void* const* d_in, const int* in_sizes, int n_in,
                              void* d_out, int out_size, void* d_ws, size_t ws_size,
                              hipStream_t stream) {
    const int*   feat_rows = (const int*)  d_in[0];
    const int*   feat_cols = (const int*)  d_in[1];
    const float* feat_vals = (const float*)d_in[2];
    const int*   adj_rows  = (const int*)  d_in[3];
    const int*   adj_cols  = (const int*)  d_in[4];
    const float* adj_vals  = (const float*)d_in[5];
    const float* W         = (const float*)d_in[6];
    float*       out       = (float*)d_out;

    const int nnz_feat = in_sizes[0];
    const int n_edges  = in_sizes[3];

    // Workspace layout:
    //   [0, XF)          : dense X f32 (204.8 MB)  -> reused as acc f32 after GEMM
    //   [XF, XF+HB)      : H f32 (102.4 MB)
    //   [.., +XB)        : Xb bf16 (102.4 MB)
    //   [.., +WT)        : Wtb bf16 (256 KB)
    const size_t XF = (size_t)NODES * IN_DIM  * sizeof(float);
    const size_t HB = (size_t)NODES * OUT_DIM * sizeof(float);
    const size_t XB = (size_t)NODES * IN_DIM  * sizeof(unsigned short);
    float*          X   = (float*)d_ws;
    float*          H   = (float*)((char*)d_ws + XF);
    unsigned short* Xb  = (unsigned short*)((char*)d_ws + XF + HB);
    unsigned short* Wtb = (unsigned short*)((char*)d_ws + XF + HB + XB);
    float*          acc = X;  // alias: X f32 dead after convert_x

    // 0) zero dense X
    {
        long long n4 = (long long)NODES * IN_DIM / 4;
        zero_ws_kernel<<<(unsigned)((n4 + 255) / 256), 256, 0, stream>>>((uint4*)X, n4);
    }
    // 1) transpose + convert W
    transpose_w_kernel<<<(IN_DIM * OUT_DIM + 255) / 256, 256, 0, stream>>>(W, Wtb);
    // 2) scatter sparse features into X (f32 atomics)
    scatter_feat_kernel<<<(nnz_feat + 255) / 256, 256, 0, stream>>>(
        feat_rows, feat_cols, feat_vals, X, nnz_feat);
    // 3) Xb = bf16(X)
    {
        long long n8 = (long long)NODES * IN_DIM / 8;
        convert_x_bf16_kernel<<<(unsigned)((n8 + 255) / 256), 256, 0, stream>>>(
            X, (uint4*)Xb, n8);
    }
    // 4) H = Xb @ W  (WMMA bf16, async LDS staging, 32-row blocks)
    spmm1_wmma_kernel<<<NODES / 32, 256, 0, stream>>>(Xb, Wtb, H);
    // 5) zero acc (aliases X region; X dead)
    {
        long long n4 = (long long)NODES * OUT_DIM / 4;
        zero_ws_kernel<<<(unsigned)((n4 + 255) / 256), 256, 0, stream>>>((uint4*)acc, n4);
    }
    // 6) acc = A @ H
    scatter_edges_kernel<<<(n_edges + 7) / 8, 256, 0, stream>>>(
        adj_rows, adj_cols, adj_vals, H, acc, n_edges);
    // 7) out = relu(acc)
    {
        long long n = (long long)out_size;
        relu_kernel<<<(unsigned)((n + 255) / 256), 256, 0, stream>>>(acc, out, n);
    }
}
